// MultiHeadAttention_29265907155033
// MI455X (gfx1250) — matrix operand order repaired and checked by reading, not compiled, not run on hardware
//
#include <hip/hip_runtime.h>
#include <math.h>

typedef __attribute__((ext_vector_type(16))) __bf16 v16bf;
typedef __attribute__((ext_vector_type(8)))  __bf16 v8bf;
typedef __attribute__((ext_vector_type(4)))  __bf16 v4bf;
typedef __attribute__((ext_vector_type(8)))  float  v8f;

#define BB 4
#define SS 2048
#define DD 1024
#define HH 16
#define DK 64
#define QK_SCALE 0.125f   // 1/sqrt(64)

// ---------------------------------------------------------------------------
// Kernel 0: fp32 -> bf16 conversion (4-wide)
// ---------------------------------------------------------------------------
__global__ void cvt_f32_bf16(const float* __restrict__ in,
                             __bf16* __restrict__ out, int n4) {
  int i = blockIdx.x * blockDim.x + threadIdx.x;
  if (i < n4) {
    float4 v = ((const float4*)in)[i];
    v4bf o;
    o[0] = (__bf16)v.x; o[1] = (__bf16)v.y;
    o[2] = (__bf16)v.z; o[3] = (__bf16)v.w;
    *(v4bf*)(out + 4 * (size_t)i) = o;
  }
}

// ---------------------------------------------------------------------------
// Kernel 1: projection GEMM  Y = X @ W^T (+bias), bf16 WMMA, head-major store
//   X: [B*S][D] bf16 row-major,  W: [D][D] bf16 row-major (row = out feature)
//   v_transposed==0 : Y[b][h][s][dk]   (for Q, K)
//   v_transposed==1 : Y[b][h][dk][s]   (for V, so PV B-frags are contiguous)
// Block = 256 thr (8 waves) computes a 64x64 output tile; wave -> 2 subtiles.
// ---------------------------------------------------------------------------
__global__ __launch_bounds__(256)
void proj_gemm(const __bf16* __restrict__ X, const __bf16* __restrict__ W,
               const float* __restrict__ bias, __bf16* __restrict__ Y,
               int v_transposed) {
  const int lane = threadIdx.x & 31;
  const int wave = threadIdx.x >> 5;
  const int half = lane >> 4;
  const int l16  = lane & 15;

  const int m0 = blockIdx.x * 64 + (wave >> 1) * 16;  // row-tile base
  const int n0 = blockIdx.y * 64 + (wave & 1) * 32;   // this wave: cols n0..n0+31

  v8f acc0 = {}; v8f acc1 = {};

  const __bf16* arow  = X + (size_t)(m0 + l16) * DD;         // A: row l16
  const __bf16* bcol0 = W + (size_t)(n0 + l16) * DD;         // B: col l16
  const __bf16* bcol1 = W + (size_t)(n0 + 16 + l16) * DD;    // B: col 16+l16

  for (int k0 = 0; k0 < DD; k0 += 32) {
    v8bf a_lo = *(const v8bf*)(arow + k0 + half * 16);
    v8bf a_hi = *(const v8bf*)(arow + k0 + half * 16 + 8);
    v16bf a;
#pragma unroll
    for (int i = 0; i < 8; ++i) { a[i] = a_lo[i]; a[8 + i] = a_hi[i]; }
    v16bf b0 = *(const v16bf*)(bcol0 + k0 + half * 16);
    v16bf b1 = *(const v16bf*)(bcol1 + k0 + half * 16);
    acc0 = __builtin_amdgcn_wmma_f32_16x16x32_bf16(false, a, false, b0,
                                                   (short)0, acc0, false, false);
    acc1 = __builtin_amdgcn_wmma_f32_16x16x32_bf16(false, a, false, b1,
                                                   (short)0, acc1, false, false);
  }

  auto store_tile = [&](const v8f& acc, int nbase) {
    const int n  = nbase + l16;
    const int h  = n >> 6, dk = n & (DK - 1);
    const float badd = bias ? bias[n] : 0.0f;
#pragma unroll
    for (int r = 0; r < 8; ++r) {
      const int m = m0 + r + half * 8;
      const int b = m >> 11, s = m & (SS - 1);
      size_t idx;
      if (v_transposed) idx = ((size_t)(b * HH + h) * DK + dk) * SS + s;
      else              idx = ((size_t)(b * HH + h) * SS + s) * DK + dk;
      Y[idx] = (__bf16)(acc[r] + badd);
    }
  };
  store_tile(acc0, n0);
  store_tile(acc1, n0 + 16);
}

// ---------------------------------------------------------------------------
// Kernel 2: full softmax strip in LDS (CDNA5: 320KB/WGP -> 128KB strip fits).
//   Phase 1: 8 waves WMMA all 128 col-tiles of a 16-row strip -> LDS (masked,
//            scaled scores).
//   Phase 2: exact row softmax: max-reduce, e=exp(v-M) in place + sum.
//   Phase 3: normalize and write fp32 attn [B][i][j][H] exactly once.
// grid = (B*H, S/16); block = 256.
// ---------------------------------------------------------------------------
__global__ __launch_bounds__(256)
void scores_softmax(const __bf16* __restrict__ Qh, const __bf16* __restrict__ Kh,
                    const unsigned char* __restrict__ mask,
                    float* __restrict__ attn) {        // [B][S][S][H] normalized
  __shared__ float sc[16][SS];     // 128 KB score strip
  __shared__ float sinv[16];       // 1/rowsum

  const int lane = threadIdx.x & 31;
  const int wave = threadIdx.x >> 5;
  const int half = lane >> 4;
  const int l16  = lane & 15;
  const int bh   = blockIdx.x;
  const int b    = bh >> 4, h = bh & 15;
  const int i0   = blockIdx.y * 16;
  const float NEG_INF = -__builtin_inff();

  // ---- Phase 1: scores into LDS ----
  // A fragments: Q rows i0..i0+15, K-dim = dk (64) -> two 16x32 chunks
  const __bf16* qrow = Qh + ((size_t)bh * SS + i0 + l16) * DK;
  v16bf a0, a1;
  {
    v8bf t0 = *(const v8bf*)(qrow + half * 16);
    v8bf t1 = *(const v8bf*)(qrow + half * 16 + 8);
    v8bf t2 = *(const v8bf*)(qrow + 32 + half * 16);
    v8bf t3 = *(const v8bf*)(qrow + 32 + half * 16 + 8);
#pragma unroll
    for (int i = 0; i < 8; ++i) {
      a0[i] = t0[i]; a0[8 + i] = t1[i];
      a1[i] = t2[i]; a1[8 + i] = t3[i];
    }
  }

  for (int t = 0; t < 16; ++t) {
    const int j0 = (wave + 8 * t) * 16;
    const __bf16* kcol = Kh + ((size_t)bh * SS + j0 + l16) * DK;
    v16bf b0 = *(const v16bf*)(kcol + half * 16);        // dk 0..31
    v16bf b1 = *(const v16bf*)(kcol + 32 + half * 16);   // dk 32..63
    v8f acc = {};
    acc = __builtin_amdgcn_wmma_f32_16x16x32_bf16(false, a0, false, b0,
                                                  (short)0, acc, false, false);
    acc = __builtin_amdgcn_wmma_f32_16x16x32_bf16(false, a1, false, b1,
                                                  (short)0, acc, false, false);
#pragma unroll
    for (int r = 0; r < 8; ++r) {
      const int i = i0 + r + half * 8;
      const int j = j0 + l16;
      const unsigned char mk = mask[(size_t)i * SS + j];
      sc[r + half * 8][j] = mk ? acc[r] * QK_SCALE : NEG_INF;
    }
  }
  __syncthreads();

  // ---- Phase 2: exact row softmax stats; e = exp(v - M) in place ----
  {
    const int row  = threadIdx.x >> 4;   // 16 threads per row (one wave half)
    const int part = threadIdx.x & 15;
    float M = NEG_INF;
    for (int c = 0; c < SS / 16; ++c)
      M = fmaxf(M, sc[row][part + 16 * c]);
#pragma unroll
    for (int m = 1; m <= 8; m <<= 1) M = fmaxf(M, __shfl_xor(M, m, 32));

    float L = 0.0f;
    for (int c = 0; c < SS / 16; ++c) {
      const int j = part + 16 * c;
      const float v = sc[row][j];
      const float e = (v > NEG_INF) ? __expf(v - M) : 0.0f;
      sc[row][j] = e;
      L += e;
    }
#pragma unroll
    for (int m = 1; m <= 8; m <<= 1) L += __shfl_xor(L, m, 32);
    if (part == 0) sinv[row] = 1.0f / ((L > 0.0f) ? L : 1.0f);
  }
  __syncthreads();

  // ---- Phase 3: normalize and write attn once ----
  for (int rr = 0; rr < 16; ++rr) {
    const float inv = sinv[rr];
    float* dst = attn + ((size_t)(b * SS + i0 + rr) * SS) * HH + h;
    for (int c = 0; c < SS / 256; ++c) {
      const int j = c * 256 + threadIdx.x;
      dst[(size_t)j * HH] = sc[rr][j] * inv;
    }
  }
}

// ---------------------------------------------------------------------------
// Kernel 3: out = P @ Vh (bf16 WMMA over j); P read from normalized attn.
// grid = (B*H, S/128); block = 256; wave w owns row-tile i0 = 128*by + 16*w
// ---------------------------------------------------------------------------
__global__ __launch_bounds__(256)
void attn_v(const float* __restrict__ attn,        // [B][S][S][H] normalized
            const __bf16* __restrict__ Vt,         // [B*H][DK][S]
            float* __restrict__ out) {             // [B][S][D]
  const int lane = threadIdx.x & 31;
  const int wave = threadIdx.x >> 5;
  const int half = lane >> 4;
  const int l16  = lane & 15;
  const int bh   = blockIdx.x;
  const int b    = bh >> 4, h = bh & 15;
  const int i0   = blockIdx.y * 128 + wave * 16;

  const int row = i0 + l16;  // A-fragment row for this lane (both halves)
  const float* arow = attn + (size_t)(b * SS + row) * SS * HH + h;  // +j*HH
  const __bf16* vbase = Vt + (size_t)bh * DK * SS;

  v8f acc[4] = {};
  for (int k0 = 0; k0 < SS; k0 += 32) {
    v16bf a;
#pragma unroll
    for (int q = 0; q < 8; ++q) {
      a[q]     = (__bf16)arow[(size_t)(k0 + half * 16 + q) * HH];
      a[8 + q] = (__bf16)arow[(size_t)(k0 + half * 16 + 8 + q) * HH];
    }
#pragma unroll
    for (int t = 0; t < 4; ++t) {
      const __bf16* vcol = vbase + (size_t)(t * 16 + l16) * SS + k0 + half * 16;
      v16bf bfrag = *(const v16bf*)vcol;
      acc[t] = __builtin_amdgcn_wmma_f32_16x16x32_bf16(false, a, false, bfrag,
                                                       (short)0, acc[t], false, false);
    }
  }

#pragma unroll
  for (int t = 0; t < 4; ++t) {
    const int dk = t * 16 + l16;
#pragma unroll
    for (int r = 0; r < 8; ++r) {
      const int i = i0 + r + half * 8;
      out[(size_t)(b * SS + i) * DD + h * DK + dk] = acc[t][r];
    }
  }
}

// ---------------------------------------------------------------------------
extern "C" void kernel_launch(void* const* d_in, const int* in_sizes, int n_in,
                              void* d_out, int out_size, void* d_ws, size_t ws_size,
                              hipStream_t stream) {
  (void)in_sizes; (void)n_in; (void)out_size; (void)ws_size;
  const float* q  = (const float*)d_in[0];
  const float* k  = (const float*)d_in[1];
  const float* v  = (const float*)d_in[2];
  const float* Wq = (const float*)d_in[3];
  const float* Wk = (const float*)d_in[4];
  const float* Wv = (const float*)d_in[5];
  const float* bv = (const float*)d_in[6];
  const unsigned char* mask = (const unsigned char*)d_in[7];

  float* out  = (float*)d_out;
  float* attn = out + (size_t)BB * SS * DD;  // tuple: (out, attn) concatenated

  // workspace carve-up (~102 MB)
  const size_t QKV_E = (size_t)BB * SS * DD;  // 8,388,608
  const size_t W_E   = (size_t)DD * DD;       // 1,048,576
  __bf16* qb  = (__bf16*)d_ws;
  __bf16* kb  = qb + QKV_E;
  __bf16* vb  = kb + QKV_E;
  __bf16* wqb = vb + QKV_E;
  __bf16* wkb = wqb + W_E;
  __bf16* wvb = wkb + W_E;
  __bf16* qh  = wvb + W_E;
  __bf16* kh  = qh + QKV_E;
  __bf16* vt  = kh + QKV_E;

  // 0) fp32 -> bf16
  {
    const int blk = 256;
    const int n4 = (int)(QKV_E / 4), g = (n4 + blk - 1) / blk;
    cvt_f32_bf16<<<g, blk, 0, stream>>>(q, qb, n4);
    cvt_f32_bf16<<<g, blk, 0, stream>>>(k, kb, n4);
    cvt_f32_bf16<<<g, blk, 0, stream>>>(v, vb, n4);
    const int w4 = (int)(W_E / 4), gw = (w4 + blk - 1) / blk;
    cvt_f32_bf16<<<gw, blk, 0, stream>>>(Wq, wqb, w4);
    cvt_f32_bf16<<<gw, blk, 0, stream>>>(Wk, wkb, w4);
    cvt_f32_bf16<<<gw, blk, 0, stream>>>(Wv, wvb, w4);
  }

  // 1) projections (head-major; V transposed)
  dim3 pg(BB * SS / 64, DD / 64);
  proj_gemm<<<pg, 256, 0, stream>>>(qb, wqb, nullptr, qh, 0);
  proj_gemm<<<pg, 256, 0, stream>>>(kb, wkb, nullptr, kh, 0);
  proj_gemm<<<pg, 256, 0, stream>>>(vb, wvb, bv, vt, 1);

  // 2) scores + exact softmax (LDS-resident strip) -> normalized attn, once
  dim3 sg(BB * HH, SS / 16);
  scores_softmax<<<sg, 256, 0, stream>>>(qh, kh, mask, attn);

  // 3) out = P @ V
  dim3 ag(BB * HH, SS / 128);
  attn_v<<<ag, 256, 0, stream>>>(attn, vt, out);
}
